// MultiHeadSelfAttention_18485539242041
// MI455X (gfx1250) — matrix operand order repaired and checked by reading, not compiled, not run on hardware
//
#include <hip/hip_runtime.h>
#include <hip/hip_bf16.h>
#include <stdint.h>

// ---------------------------------------------------------------------------
// Types for CDNA5 WMMA / TDM (gfx1250, wave32)
// ---------------------------------------------------------------------------
typedef __bf16 bf16;
typedef __attribute__((ext_vector_type(16))) __bf16   v16bf;
typedef __attribute__((ext_vector_type(8)))  float    v8f;
typedef __attribute__((ext_vector_type(4)))  uint32_t u32x4;
typedef __attribute__((ext_vector_type(8)))  int      i32x8;
typedef __attribute__((ext_vector_type(4)))  int      i32x4;

struct alignas(16) U4 { uint32_t x0, x1, x2, x3; };   // POD 16-byte chunk

union FragBF {            // 16 bf16 = 32 bytes = 8 VGPRs (A or B fragment)
    U4    u[2];
    v16bf v;
};

#define D_MODEL   1024
#define SEQ       4096
#define NUM_HEADS 16
#define HEAD_DIM  64

#if __has_builtin(__builtin_amdgcn_tensor_load_to_lds) && \
    __has_builtin(__builtin_amdgcn_s_wait_tensorcnt)
#define HAVE_TDM 1
#else
#define HAVE_TDM 0
#endif

#if HAVE_TDM
// ---------------------------------------------------------------------------
// Issue one TDM 2D tile load: global (tile_h rows x tile_w elems, bf16,
// row stride = stride_elems) -> LDS at lds_off (rows stored contiguously).
// D# layout per CDNA5 ISA ch.8 (group0 128b, group1 256b; groups 2/3 zero
// => 2D tensor mode, tile_dim2=0 unused). Tensor dims set huge: no OOB clip.
// Toolchain uses the 6-arg builtin: (g0, g1, g2, g3, g4, cpol).
// ---------------------------------------------------------------------------
__device__ __forceinline__ void tdm_load_2d(uint32_t lds_off, const void* gaddr,
                                            uint32_t tile_w, uint32_t tile_h,
                                            uint32_t stride_elems) {
    const uint64_t ga = (uint64_t)gaddr;
    u32x4 g0;
    g0[0] = 1u;                                   // count=1 (valid user D#)
    g0[1] = lds_off;                              // LDS byte address
    g0[2] = (uint32_t)ga;                         // global_addr[31:0]
    g0[3] = (uint32_t)((ga >> 32) & 0x1FFFFFFu)   // global_addr[56:32]
          | (2u << 30);                           // type = 2 ("image")
    const uint32_t dim = 1u << 30;                // huge tensor dims
    i32x8 g1;
    g1[0] = (int)(1u << 16);                      // wg_mask=0, data_size=1 (2B)
    g1[1] = (int)((dim & 0xFFFFu) << 16);         // tensor_dim0[15:0]
    g1[2] = (int)((dim >> 16) | ((dim & 0xFFFFu) << 16));   // d0 hi | d1 lo
    g1[3] = (int)((dim >> 16) | (tile_w << 16));  // d1 hi | tile_dim0
    g1[4] = (int)(tile_h);                        // tile_dim1 | tile_dim2=0
    g1[5] = (int)stride_elems;                    // tensor_dim0_stride[31:0]
    g1[6] = 0;                                    // d0_stride hi | d1_stride lo
    g1[7] = 0;
    i32x4 gz4 = {0, 0, 0, 0};
    i32x8 gz8 = {0, 0, 0, 0, 0, 0, 0, 0};
    __builtin_amdgcn_tensor_load_to_lds(g0, g1, gz4, gz4, gz8, 0);
}
#endif

// ---------------------------------------------------------------------------
// Kernel 1: fp32 -> bf16 (vectorized by 4)
// ---------------------------------------------------------------------------
__global__ void cvt_f32_to_bf16(const float* __restrict__ in,
                                bf16* __restrict__ out, int n) {
    int i = (blockIdx.x * blockDim.x + threadIdx.x) * 4;
    if (i + 3 < n) {
        float4 f = *(const float4*)(in + i);
        out[i + 0] = (bf16)f.x;
        out[i + 1] = (bf16)f.y;
        out[i + 2] = (bf16)f.z;
        out[i + 3] = (bf16)f.w;
    }
}

// ---------------------------------------------------------------------------
// Kernel 2: W[in][out] fp32 -> Wt[out][in] bf16  (N-major weights for GEMM B)
// ---------------------------------------------------------------------------
__global__ void cvt_transpose(const float* __restrict__ W,
                              bf16* __restrict__ Wt) {
    int k = blockIdx.x * blockDim.x + threadIdx.x;  // contiguous write
    int n = blockIdx.y;
    Wt[(size_t)n * D_MODEL + k] = (bf16)W[(size_t)k * D_MODEL + n];
}

// ---------------------------------------------------------------------------
// Kernel 3: GEMM  C[M,N] = A[M,K] * Bt[N,K]^T   (A,Bt bf16, acc fp32)
// 128x128 block tile, 256 threads = 8 waves in 4(m) x 2(n) layout.
// Tiles staged via TDM (tensor_load_to_lds, TENSORcnt) when available,
// else per-thread b128 loads. Double-buffered LDS, one barrier per k-step.
// ---------------------------------------------------------------------------
template <bool OUT_BF16>
__global__ __launch_bounds__(256) void gemm_bf16_nt(
    const bf16* __restrict__ A, const bf16* __restrict__ Bt,
    bf16* __restrict__ Cb, float* __restrict__ Cf,
    int M, int N, int K) {
    __shared__ bf16 As[2][128][32];
    __shared__ bf16 Bs[2][128][32];

    const int tid  = threadIdx.x;
    const int lane = tid & 31;
    const int wave = tid >> 5;     // 0..7
    const int wm   = wave >> 1;    // 0..3 -> 32 rows each
    const int wn   = wave & 1;     // 0..1 -> 64 cols each
    const int ln   = lane & 15;
    const int hi   = lane >> 4;

    const int m0 = blockIdx.y * 128;
    const int n0 = blockIdx.x * 128;

    const v8f vzero = {0.f, 0.f, 0.f, 0.f, 0.f, 0.f, 0.f, 0.f};
    v8f acc[2][4];
#pragma unroll
    for (int i = 0; i < 2; ++i)
#pragma unroll
        for (int j = 0; j < 4; ++j) acc[i][j] = vzero;

    const int lr = tid >> 1;        // 0..127: row within tile
    const int lc = (tid & 1) * 16;  // 0 or 16: k-offset

    const bf16* ga = A  + (size_t)(m0 + lr) * K + lc;
    const bf16* gb = Bt + (size_t)(n0 + lr) * K + lc;

    // ---- prologue: stage k0 = 0 into buffer 0
#if HAVE_TDM
    const bool issuer = (wave == 0);     // TDM is wave-level; one wave issues
    if (issuer) {
        tdm_load_2d((uint32_t)(uintptr_t)&As[0][0][0],
                    A + (size_t)m0 * K, 32, 128, (uint32_t)K);
        tdm_load_2d((uint32_t)(uintptr_t)&Bs[0][0][0],
                    Bt + (size_t)n0 * K, 32, 128, (uint32_t)K);
        __builtin_amdgcn_s_wait_tensorcnt(0);
    }
#else
    {
        U4 a0 = *(const U4*)(ga);
        U4 a1 = *(const U4*)(ga + 8);
        U4 b0 = *(const U4*)(gb);
        U4 b1 = *(const U4*)(gb + 8);
        *(U4*)&As[0][lr][lc]     = a0;
        *(U4*)&As[0][lr][lc + 8] = a1;
        *(U4*)&Bs[0][lr][lc]     = b0;
        *(U4*)&Bs[0][lr][lc + 8] = b1;
    }
#endif
    __syncthreads();

    int cur = 0;
    for (int k0 = 0; k0 < K; k0 += 32) {
        const bool has_next = (k0 + 32) < K;
        const int  nxt      = cur ^ 1;

#if HAVE_TDM
        if (has_next && issuer) {          // DMA next tile during compute
            tdm_load_2d((uint32_t)(uintptr_t)&As[nxt][0][0],
                        A + (size_t)m0 * K + k0 + 32, 32, 128, (uint32_t)K);
            tdm_load_2d((uint32_t)(uintptr_t)&Bs[nxt][0][0],
                        Bt + (size_t)n0 * K + k0 + 32, 32, 128, (uint32_t)K);
        }
#else
        U4 a0, a1, b0, b1;
        if (has_next) {                    // issue next global loads first
            const bf16* na = ga + k0 + 32;
            const bf16* nb = gb + k0 + 32;
            a0 = *(const U4*)(na);
            a1 = *(const U4*)(na + 8);
            b0 = *(const U4*)(nb);
            b1 = *(const U4*)(nb + 8);
        }
#endif

        // ---- compute from buffer `cur` (overlaps in-flight tile transfer)
        FragBF afrag[2], bfrag[4];
#pragma unroll
        for (int mt = 0; mt < 2; ++mt) {
            int m = wm * 32 + mt * 16 + ln;   // A: lane = row
            afrag[mt].u[0] = *(const U4*)&As[cur][m][8 * hi];
            afrag[mt].u[1] = *(const U4*)&As[cur][m][16 + 8 * hi];
        }
#pragma unroll
        for (int nt = 0; nt < 4; ++nt) {
            int n = wn * 64 + nt * 16 + ln;   // B: lane = col
            bfrag[nt].u[0] = *(const U4*)&Bs[cur][n][8 * hi];
            bfrag[nt].u[1] = *(const U4*)&Bs[cur][n][16 + 8 * hi];
        }
#pragma unroll
        for (int mt = 0; mt < 2; ++mt)
#pragma unroll
            for (int nt = 0; nt < 4; ++nt)
                acc[mt][nt] = __builtin_amdgcn_wmma_f32_16x16x32_bf16(
                    false, afrag[mt].v, false, bfrag[nt].v,
                    (short)0, acc[mt][nt], false, false);

        if (has_next) {
#if HAVE_TDM
            if (issuer) __builtin_amdgcn_s_wait_tensorcnt(0);
#else
            *(U4*)&As[nxt][lr][lc]     = a0;
            *(U4*)&As[nxt][lr][lc + 8] = a1;
            *(U4*)&Bs[nxt][lr][lc]     = b0;
            *(U4*)&Bs[nxt][lr][lc + 8] = b1;
#endif
            __syncthreads();               // one barrier per k-step
            cur = nxt;
        }
    }

    // Epilogue: C layout -> lane = N col; VGPR v -> row v + 8*hi
#pragma unroll
    for (int mt = 0; mt < 2; ++mt)
#pragma unroll
        for (int nt = 0; nt < 4; ++nt)
#pragma unroll
            for (int v = 0; v < 8; ++v) {
                int row = m0 + wm * 32 + mt * 16 + v + 8 * hi;
                int col = n0 + wn * 64 + nt * 16 + ln;
                if constexpr (OUT_BF16)
                    Cb[(size_t)row * N + col] = (bf16)acc[mt][nt][v];
                else
                    Cf[(size_t)row * N + col] = acc[mt][nt][v];
            }
}

// ---------------------------------------------------------------------------
// Kernel 4: causal flash attention per (qblock=64 rows, head).
// 128 threads = 4 waves; wave w owns query rows [w*16, w*16+16).
// ---------------------------------------------------------------------------
__global__ __launch_bounds__(128) void attn_causal(
    const bf16* __restrict__ Qg, const bf16* __restrict__ Kg,
    const bf16* __restrict__ Vg, bf16* __restrict__ Og) {
    __shared__ bf16 Qs[64][64];
    __shared__ bf16 Ks[64][64];   // [key][hd]
    __shared__ bf16 Vt[64][64];   // [hd][key]   (transposed on load)
    __shared__ bf16 Ps[64][64];   // probabilities, [q][key]

    const int tid  = threadIdx.x;
    const int lane = tid & 31;
    const int wave = tid >> 5;    // 0..3
    const int ln   = lane & 15;
    const int hi   = lane >> 4;

    const int qb = blockIdx.x;           // 0..63
    const int h  = blockIdx.y;           // 0..15
    const int q0 = qb * 64;
    const size_t ld   = D_MODEL;
    const size_t hoff = (size_t)h * HEAD_DIM;

    const int lr = tid >> 1;             // tile row handled by this thread
    const int lc = (tid & 1) * 32;       // 32-elem column chunk

    // Load Q tile: 64x64 bf16, 32 elems / thread
    {
        const bf16* g = Qg + (size_t)(q0 + lr) * ld + hoff + lc;
#pragma unroll
        for (int j = 0; j < 4; ++j)
            *(U4*)&Qs[lr][lc + 8 * j] = *(const U4*)(g + 8 * j);
    }

    float run_max[8], run_sum[8];
    const v8f vzero = {0.f, 0.f, 0.f, 0.f, 0.f, 0.f, 0.f, 0.f};
    v8f oacc[4];
#pragma unroll
    for (int v = 0; v < 8; ++v) { run_max[v] = -3.0e38f; run_sum[v] = 0.f; }
#pragma unroll
    for (int t = 0; t < 4; ++t) oacc[t] = vzero;

    for (int kb = 0; kb <= qb; ++kb) {
        const int kk0 = kb * 64;
        __syncthreads();   // protect Ks/Vt readers of previous iteration
        {
            const bf16* gk = Kg + (size_t)(kk0 + lr) * ld + hoff + lc;
#pragma unroll
            for (int j = 0; j < 4; ++j)
                *(U4*)&Ks[lr][lc + 8 * j] = *(const U4*)(gk + 8 * j);
            const bf16* gv = Vg + (size_t)(kk0 + lr) * ld + hoff + lc;
#pragma unroll
            for (int j = 0; j < 4; ++j) {
                U4 d = *(const U4*)(gv + 8 * j);
                const bf16* e = (const bf16*)&d;
#pragma unroll
                for (int x = 0; x < 8; ++x) Vt[lc + 8 * j + x][lr] = e[x];
            }
        }
        // prefetch next key-block's K/V rows (global_prefetch_b8)
        if (kb < qb) {
            __builtin_prefetch(Kg + (size_t)(kk0 + 64 + lr) * ld + hoff + lc, 0, 1);
            __builtin_prefetch(Vg + (size_t)(kk0 + 64 + lr) * ld + hoff + lc, 0, 1);
        }
        __syncthreads();

        // ---- S = Q * K^T  (per wave: 16 x 64), K-dim = HEAD_DIM = 2 k-steps
        v8f sacc[4];
#pragma unroll
        for (int t = 0; t < 4; ++t) sacc[t] = vzero;
#pragma unroll
        for (int ks = 0; ks < 2; ++ks) {
            FragBF af, bfrag[4];
            int m = wave * 16 + ln;
            af.u[0] = *(const U4*)&Qs[m][ks * 32 + 8 * hi];
            af.u[1] = *(const U4*)&Qs[m][ks * 32 + 16 + 8 * hi];
#pragma unroll
            for (int t = 0; t < 4; ++t) {      // batch all B-frag loads first
                int n = t * 16 + ln;           // key index (B col)
                bfrag[t].u[0] = *(const U4*)&Ks[n][ks * 32 + 8 * hi];
                bfrag[t].u[1] = *(const U4*)&Ks[n][ks * 32 + 16 + 8 * hi];
            }
#pragma unroll
            for (int t = 0; t < 4; ++t)
                sacc[t] = __builtin_amdgcn_wmma_f32_16x16x32_bf16(
                    false, af.v, false, bfrag[t].v, (short)0, sacc[t],
                    false, false);
        }

        // ---- online softmax per row (row = q0 + wave*16 + v + 8*hi)
#pragma unroll
        for (int v = 0; v < 8; ++v) {
            int row = q0 + wave * 16 + v + 8 * hi;
            float mx = -3.0e38f;
            float sc[4];
#pragma unroll
            for (int t = 0; t < 4; ++t) {
                int col = kk0 + t * 16 + ln;
                float s = sacc[t][v] * 0.125f;        // 1/sqrt(64)
                s = (col <= row) ? s : -3.0e38f;      // causal mask
                sc[t] = s;
                mx = fmaxf(mx, s);
            }
#pragma unroll
            for (int off = 1; off < 16; off <<= 1)    // stays in half-wave
                mx = fmaxf(mx, __shfl_xor(mx, off, 32));
            float mnew  = fmaxf(run_max[v], mx);
            float scale = __expf(run_max[v] - mnew);
            float psum  = 0.f;
#pragma unroll
            for (int t = 0; t < 4; ++t) {
                float p = __expf(sc[t] - mnew);
                psum += p;
                Ps[wave * 16 + v + 8 * hi][t * 16 + ln] = (bf16)p;
            }
#pragma unroll
            for (int off = 1; off < 16; off <<= 1)
                psum += __shfl_xor(psum, off, 32);
            run_sum[v] = run_sum[v] * scale + psum;
            run_max[v] = mnew;
#pragma unroll
            for (int t = 0; t < 4; ++t) oacc[t][v] *= scale;
        }
        // No barrier: Ps rows are wave-local and same-wave LDS ops are
        // in-order (DScnt); the compiler orders the dependent ds_loads.

        // ---- O += P * V   (A = Ps rows of this wave; B = Vt, N = head dim)
#pragma unroll
        for (int ks = 0; ks < 2; ++ks) {
            FragBF af, bfrag[4];
            int m = wave * 16 + ln;
            af.u[0] = *(const U4*)&Ps[m][ks * 32 + 8 * hi];
            af.u[1] = *(const U4*)&Ps[m][ks * 32 + 16 + 8 * hi];
#pragma unroll
            for (int t = 0; t < 4; ++t) {      // batch all B-frag loads first
                int n = t * 16 + ln;           // head-dim col
                bfrag[t].u[0] = *(const U4*)&Vt[n][ks * 32 + 8 * hi];
                bfrag[t].u[1] = *(const U4*)&Vt[n][ks * 32 + 16 + 8 * hi];
            }
#pragma unroll
            for (int t = 0; t < 4; ++t)
                oacc[t] = __builtin_amdgcn_wmma_f32_16x16x32_bf16(
                    false, af.v, false, bfrag[t].v, (short)0, oacc[t],
                    false, false);
        }
    }

    // ---- epilogue: normalize and store bf16
#pragma unroll
    for (int t = 0; t < 4; ++t)
#pragma unroll
        for (int v = 0; v < 8; ++v) {
            int row = q0 + wave * 16 + v + 8 * hi;
            int col = t * 16 + ln;
            float o = oacc[t][v] / run_sum[v];
            Og[(size_t)row * ld + hoff + col] = (bf16)o;
        }
}

// ---------------------------------------------------------------------------
// Host-side orchestration
// ---------------------------------------------------------------------------
extern "C" void kernel_launch(void* const* d_in, const int* in_sizes, int n_in,
                              void* d_out, int out_size, void* d_ws,
                              size_t ws_size, hipStream_t stream) {
    (void)in_sizes; (void)n_in; (void)out_size; (void)ws_size;

    const float* x  = (const float*)d_in[0];
    const float* Wq = (const float*)d_in[1];
    const float* Wk = (const float*)d_in[2];
    const float* Wv = (const float*)d_in[3];
    const float* Wo = (const float*)d_in[4];
    float* out = (float*)d_out;

    char* ws = (char*)d_ws;
    const size_t MB = 1024 * 1024;
    bf16* xb  = (bf16*)(ws + 0);        //  8 MiB: x bf16
    bf16* WqT = (bf16*)(ws + 8  * MB);  //  2 MiB each: transposed bf16 weights
    bf16* WkT = (bf16*)(ws + 10 * MB);
    bf16* WvT = (bf16*)(ws + 12 * MB);
    bf16* WoT = (bf16*)(ws + 14 * MB);
    bf16* Qb  = (bf16*)(ws + 16 * MB);  //  8 MiB
    bf16* Kb  = (bf16*)(ws + 24 * MB);  //  8 MiB
    bf16* Vb  = (bf16*)(ws + 32 * MB);  //  8 MiB
    bf16* AO  = (bf16*)(ws + 40 * MB);  //  8 MiB attention output

    // 1) conversions
    cvt_f32_to_bf16<<<(SEQ * D_MODEL / 4 + 255) / 256, 256, 0, stream>>>(
        x, xb, SEQ * D_MODEL);
    dim3 tg(D_MODEL / 256, D_MODEL);
    cvt_transpose<<<tg, 256, 0, stream>>>(Wq, WqT);
    cvt_transpose<<<tg, 256, 0, stream>>>(Wk, WkT);
    cvt_transpose<<<tg, 256, 0, stream>>>(Wv, WvT);
    cvt_transpose<<<tg, 256, 0, stream>>>(Wo, WoT);

    // 2) QKV projections (WMMA bf16 GEMMs, TDM-staged tiles)
    dim3 gg(D_MODEL / 128, SEQ / 128);
    gemm_bf16_nt<true><<<gg, 256, 0, stream>>>(xb, WqT, Qb, (float*)nullptr,
                                               SEQ, D_MODEL, D_MODEL);
    gemm_bf16_nt<true><<<gg, 256, 0, stream>>>(xb, WkT, Kb, (float*)nullptr,
                                               SEQ, D_MODEL, D_MODEL);
    gemm_bf16_nt<true><<<gg, 256, 0, stream>>>(xb, WvT, Vb, (float*)nullptr,
                                               SEQ, D_MODEL, D_MODEL);

    // 3) causal flash attention
    dim3 ag(SEQ / 64, NUM_HEADS);
    attn_causal<<<ag, 128, 0, stream>>>(Qb, Kb, Vb, AO);

    // 4) output projection -> fp32 d_out
    gemm_bf16_nt<false><<<gg, 256, 0, stream>>>(AO, WoT, (bf16*)nullptr, out,
                                                SEQ, D_MODEL, D_MODEL);
}
